// TriplaneSdf_39582418600324
// MI455X (gfx1250) — matrix operand order repaired and checked by reading, not compiled, not run on hardware
//
#include <hip/hip_runtime.h>
#include <math.h>

typedef __attribute__((ext_vector_type(16))) _Float16 v16h;
typedef __attribute__((ext_vector_type(8)))  _Float16 v8h;
typedef __attribute__((ext_vector_type(8)))  float    v8f;

#define BLK 64            // 2 waves of 32; each wave owns 32 points
#define FS  104           // feat row stride in halves (208B, 16B aligned)
#define HS  40            // hidden row stride in halves (80B, 16B aligned)
#define RESOL 512
#define PLANE_CH_STRIDE (RESOL*RESOL)   // elements between channels

// single-instruction relu / clamp via v_med3_num_f32
__device__ __forceinline__ float relu1(float x) {
  return __builtin_amdgcn_fmed3f(x, 0.0f, 3.0e38f);
}
__device__ __forceinline__ float clampf3(float x, float lo, float hi) {
  return __builtin_amdgcn_fmed3f(x, lo, hi);
}

// ---------------------------------------------------------------------------
// A-fragment loader: 16x32 f16 tile from LDS (row-major halves)
// lanes 0-15 -> row M=lane,    halves 0..15 = K 0..15
// lanes 16-31 -> row M=lane-16, halves 0..15 = K 16..31
// ---------------------------------------------------------------------------
__device__ __forceinline__ v16h load_a16x32(const _Float16* base, int rowBase,
                                            int strideHalves, int kOffHalves) {
  const int lane = threadIdx.x & 31;
  const int m    = lane & 15;
  const int kb   = (lane < 16) ? 0 : 16;
  const _Float16* p = base + (rowBase + m) * strideHalves + kOffHalves + kb;
  union { v16h v; v8h h[2]; } u;
  u.h[0] = *(const v8h*)(p);        // ds_load_b128
  u.h[1] = *(const v8h*)(p + 8);    // ds_load_b128
  return u.v;
}

// ---------------------------------------------------------------------------
// Pre-swizzled B fragments: for each (kt,nt) fragment, 32 lanes x 16 halves
// contiguous -> each lane reads its 32B with two ds_load_b128.
// Fragment semantics (32x16 f16 B): col N = lane%16,
//   lanes 0-15: halves h -> K = kt*32 + h ; lanes 16-31: K = kt*32 + 16 + h
// ---------------------------------------------------------------------------
__device__ __forceinline__ v16h load_b_swz(const _Float16* buf, int kt, int nt) {
  const int lane = threadIdx.x & 31;
  const _Float16* p = buf + (((kt << 1) | nt) * 32 + lane) * 16;
  union { v16h v; v8h h[2]; } u;
  u.h[0] = *(const v8h*)(p);
  u.h[1] = *(const v8h*)(p + 8);
  return u.v;
}

// Cooperative fill of the swizzled buffer from a row-major (K x 32) f32 matrix.
__device__ __forceinline__ void fill_b_swz(const float* __restrict__ w,
                                           _Float16* buf, int KT, int tid) {
  const int nChunks = KT * 2 * 32;            // one 16-half chunk per (frag,lane)
  for (int chunk = tid; chunk < nChunks; chunk += BLK) {
    const int lane_s = chunk & 31;
    const int frag   = chunk >> 5;            // kt*2 + nt
    const int nt     = frag & 1;
    const int kt     = frag >> 1;
    const int n      = lane_s & 15;
    const int kb     = (lane_s < 16) ? 0 : 16;
    const float* src = w + (kt * 32 + kb) * 32 + nt * 16 + n;
    v8h lo, hi;
#pragma unroll
    for (int h = 0; h < 8; ++h) lo[h] = (_Float16)src[h * 32];
#pragma unroll
    for (int h = 0; h < 8; ++h) hi[h] = (_Float16)src[(h + 8) * 32];
    _Float16* d = buf + chunk * 16;
    *(v8h*)(d)     = lo;   // ds_store_b128
    *(v8h*)(d + 8) = hi;   // ds_store_b128
  }
}

// ---------------------------------------------------------------------------
// Bilinear gather of 32 channels from one (32,512,512) plane; packed stores.
// ---------------------------------------------------------------------------
__device__ __forceinline__ void sample_plane(const float* __restrict__ pl,
                                             float gx, float gy, _Float16* dst) {
  const float Wm1 = (float)(RESOL - 1);
  float ix = clampf3((gx + 1.0f) * 0.5f * Wm1, 0.0f, Wm1);
  float iy = clampf3((gy + 1.0f) * 0.5f * Wm1, 0.0f, Wm1);
  float ix0f = floorf(ix), iy0f = floorf(iy);
  float wx = ix - ix0f,    wy = iy - iy0f;
  int ix0 = (int)ix0f,     iy0 = (int)iy0f;
  int ix1 = min(ix0 + 1, RESOL - 1);
  int iy1 = min(iy0 + 1, RESOL - 1);
  float w00 = (1.0f - wy) * (1.0f - wx);
  float w01 = (1.0f - wy) * wx;
  float w10 = wy * (1.0f - wx);
  float w11 = wy * wx;
  const float* p00 = pl + (size_t)iy0 * RESOL + ix0;
  const float* p01 = pl + (size_t)iy0 * RESOL + ix1;
  const float* p10 = pl + (size_t)iy1 * RESOL + ix0;
  const float* p11 = pl + (size_t)iy1 * RESOL + ix1;
#pragma unroll
  for (int cg = 0; cg < 4; ++cg) {           // 4 groups of 8 channels
    v8h pack;
#pragma unroll
    for (int c8 = 0; c8 < 8; ++c8) {
      const size_t off = (size_t)(cg * 8 + c8) * PLANE_CH_STRIDE;
      float v = w00 * p00[off] + w01 * p01[off] + w10 * p10[off] + w11 * p11[off];
      pack[c8] = (_Float16)v;
    }
    *(v8h*)(dst + cg * 8) = pack;            // ds_store_b128
  }
}

__global__ __launch_bounds__(BLK)
void TriplaneSdf_kernel(const float* __restrict__ pos,
                        const float* __restrict__ pxy, const float* __restrict__ pxz,
                        const float* __restrict__ pyz,
                        const float* __restrict__ wsdf0, const float* __restrict__ wsdf1,
                        const float* __restrict__ wrgb0, const float* __restrict__ wrgb1,
                        const float* __restrict__ wrgb2,
                        float* __restrict__ out) {
  __shared__ __align__(16) _Float16 sFeat[BLK * FS];   // 32x96 f16 per wave
  __shared__ __align__(16) _Float16 sHA[BLK * HS];     // hidden 32x32 per wave
  __shared__ __align__(16) _Float16 sHB[BLK * HS];     // hidden 32x32 per wave
  __shared__ __align__(16) _Float16 sW0s[96 * 32];     // swizzled B frags
  __shared__ __align__(16) _Float16 sW0r[96 * 32];     // swizzled B frags
  __shared__ __align__(16) _Float16 sW1r[32 * 32];     // swizzled B frags
  __shared__ float sDist[BLK];

  const int tid     = threadIdx.x;
  const int lane    = tid & 31;
  const int wave    = tid >> 5;
  const int rowBase = wave * 32;
  const long long blockBase = (long long)blockIdx.x * BLK;

  // ---- stage 0: weights -> LDS f16, pre-swizzled into B-fragment order ----
  fill_b_swz(wsdf0, sW0s, 3, tid);
  fill_b_swz(wrgb0, sW0r, 3, tid);
  fill_b_swz(wrgb1, sW1r, 1, tid);

  // ---- stage 1: per-thread triplane gather -> feat row in LDS ----
  {
    const long long i = blockBase + tid;
    float x = pos[i * 3 + 0];
    float y = pos[i * 3 + 1];
    float z = pos[i * 3 + 2];
    sDist[tid] = sqrtf(x * x + y * y + z * z);
    const float inv = 2.0f;              // 1/RADIUS
    float p0 = x * inv, p1 = y * inv, p2 = z * inv;
    _Float16* frow = &sFeat[tid * FS];
    sample_plane(pxy, p1, p0, frow + 0);   // f_xy(gx=p1, gy=p0)
    sample_plane(pxz, p2, p0, frow + 32);  // f_xz(gx=p2, gy=p0)
    sample_plane(pyz, p2, p1, frow + 64);  // f_yz(gx=p2, gy=p1)
  }
  __syncthreads();

  // ---- cache A fragments of feat (2 M-tiles x 3 K-tiles) ----
  v16h aF[2][3];
#pragma unroll
  for (int mt = 0; mt < 2; ++mt)
#pragma unroll
    for (int kt = 0; kt < 3; ++kt)
      aF[mt][kt] = load_a16x32(sFeat, rowBase + mt * 16, FS, kt * 32);

  const int nn   = lane & 15;
  const int moff = (lane >= 16) ? 8 : 0;

  // layer: relu(feat(32x96) @ W(96x32)) -> H (LDS f16); B frags hoisted over mt
  auto layer96 = [&](const _Float16* Wswz, _Float16* H) {
#pragma unroll
    for (int nt = 0; nt < 2; ++nt) {
      const v16h b0 = load_b_swz(Wswz, 0, nt);
      const v16h b1 = load_b_swz(Wswz, 1, nt);
      const v16h b2 = load_b_swz(Wswz, 2, nt);
#pragma unroll
      for (int mt = 0; mt < 2; ++mt) {
        v8f acc = {};
        acc = __builtin_amdgcn_wmma_f32_16x16x32_f16(false, aF[mt][0], false, b0,
                                                     (short)0, acc, false, false);
        acc = __builtin_amdgcn_wmma_f32_16x16x32_f16(false, aF[mt][1], false, b1,
                                                     (short)0, acc, false, false);
        acc = __builtin_amdgcn_wmma_f32_16x16x32_f16(false, aF[mt][2], false, b2,
                                                     (short)0, acc, false, false);
#pragma unroll
        for (int r = 0; r < 8; ++r)
          H[(rowBase + mt * 16 + moff + r) * HS + nt * 16 + nn] =
              (_Float16)relu1(acc[r]);
      }
    }
  };

  // ---- sdf branch: h_sdf = relu(feat@Wsdf0); sdf_pre = h_sdf @ [wsdf1 | 0] ----
  layer96(sW0s, sHA);
  v16h bs;
  {
    const int kb = (lane < 16) ? 0 : 16;
#pragma unroll
    for (int h = 0; h < 16; ++h)
      bs[h] = (nn == 0) ? (_Float16)wsdf1[kb + h] : (_Float16)0.0f;
  }
  v8f accS[2];
#pragma unroll
  for (int mt = 0; mt < 2; ++mt) {
    v16h a = load_a16x32(sHA, rowBase + mt * 16, HS, 0);
    v8f z = {};
    accS[mt] = __builtin_amdgcn_wmma_f32_16x16x32_f16(false, a, false, bs,
                                                      (short)0, z, false, false);
  }

  // ---- rgb branch: h_rgb = relu(feat@Wrgb0) (reuse sHA, same-wave LDS order) ----
  layer96(sW0r, sHA);

  // h2 = relu(h_rgb @ Wrgb1) -> sHB
#pragma unroll
  for (int mt = 0; mt < 2; ++mt) {
    v16h a = load_a16x32(sHA, rowBase + mt * 16, HS, 0);
#pragma unroll
    for (int nt = 0; nt < 2; ++nt) {
      v16h b = load_b_swz(sW1r, 0, nt);
      v8f z = {};
      v8f acc = __builtin_amdgcn_wmma_f32_16x16x32_f16(false, a, false, b,
                                                       (short)0, z, false, false);
#pragma unroll
      for (int r = 0; r < 8; ++r)
        sHB[(rowBase + mt * 16 + moff + r) * HS + nt * 16 + nn] =
            (_Float16)relu1(acc[r]);
    }
  }

  // rgb_pre = h2 @ [wrgb2 | 0]  (32x16, cols 0..2 valid)
  v16h br;
  {
    const int kb = (lane < 16) ? 0 : 16;
#pragma unroll
    for (int h = 0; h < 16; ++h)
      br[h] = (nn < 3) ? (_Float16)wrgb2[(kb + h) * 3 + nn] : (_Float16)0.0f;
  }
  v8f accR[2];
#pragma unroll
  for (int mt = 0; mt < 2; ++mt) {
    v16h a = load_a16x32(sHB, rowBase + mt * 16, HS, 0);
    v8f z = {};
    accR[mt] = __builtin_amdgcn_wmma_f32_16x16x32_f16(false, a, false, br,
                                                      (short)0, z, false, false);
  }

  // ---- epilogue: column-0 lanes write sdf, columns 0..2 write r, g, b ----
  // tanh via native v_exp/v_rcp: tanh(x) = (e^{2x}-1) * rcp(e^{2x}+1)
  // sigmoid(x) = rcp(1 + e^{-x})
#pragma unroll
  for (int mt = 0; mt < 2; ++mt) {
#pragma unroll
    for (int r = 0; r < 8; ++r) {
      const int row = rowBase + mt * 16 + moff + r;
      const long long pt = blockBase + row;
      if (nn == 0) {
        float x2 = clampf3(2.0f * accS[mt][r], -20.0f, 20.0f);
        float e  = __expf(x2);
        float th = (e - 1.0f) * __builtin_amdgcn_rcpf(e + 1.0f);
        // tanh(s)*RADIUS*1.732*1.5 + (dist - 0.1*RADIUS)
        out[pt * 4 + 0] = th * 1.2990381057f + sDist[row] - 0.05f;
      }
      if (nn < 3) {
        float e = __expf(-accR[mt][r]);
        out[pt * 4 + 1 + nn] = __builtin_amdgcn_rcpf(1.0f + e);
      }
    }
  }
}

extern "C" void kernel_launch(void* const* d_in, const int* in_sizes, int n_in,
                              void* d_out, int out_size, void* d_ws, size_t ws_size,
                              hipStream_t stream) {
  const float* pos   = (const float*)d_in[0];
  const float* pxy   = (const float*)d_in[1];
  const float* pxz   = (const float*)d_in[2];
  const float* pyz   = (const float*)d_in[3];
  const float* wsdf0 = (const float*)d_in[4];
  const float* wsdf1 = (const float*)d_in[5];
  const float* wrgb0 = (const float*)d_in[6];
  const float* wrgb1 = (const float*)d_in[7];
  const float* wrgb2 = (const float*)d_in[8];
  float* out = (float*)d_out;

  const int n    = in_sizes[0] / 3;        // number of points
  const int grid = (n + BLK - 1) / BLK;    // N = 2^20 -> divisible by BLK
  hipLaunchKernelGGL(TriplaneSdf_kernel, dim3(grid), dim3(BLK), 0, stream,
                     pos, pxy, pxz, pyz, wsdf0, wsdf1, wrgb0, wrgb1, wrgb2, out);
}
